// SuperBlock_65206193488489
// MI455X (gfx1250) — matrix operand order repaired and checked by reading, not compiled, not run on hardware
//
#include <hip/hip_runtime.h>

// ---------------------------------------------------------------------------
// MI455X / gfx1250 fused Transformer-XL block (4 steps).
// f16 operands everywhere (converted once), f32 WMMA accumulation.
// Async global->LDS staging (double-buffered, partial s_wait_asynccnt so the
// async engine overlaps the WMMAs); fragments = wide ds_load_b128 pairs.
// ---------------------------------------------------------------------------

typedef __attribute__((ext_vector_type(16))) _Float16 v16h;
typedef __attribute__((ext_vector_type(8)))  _Float16 h8;
typedef __attribute__((ext_vector_type(8)))  float    v8f;
typedef __attribute__((ext_vector_type(4)))  int      v4i;

__device__ __forceinline__ v8f wmma16(v16h a, v16h b, v8f c) {
  return __builtin_amdgcn_wmma_f32_16x16x32_f16(false, a, false, b, (short)0, c,
                                                false, false);
}
__device__ __forceinline__ v16h frag_cat(h8 lo, h8 hi) {
  return __builtin_shufflevector(lo, hi, 0, 1, 2, 3, 4, 5, 6, 7,
                                 8, 9, 10, 11, 12, 13, 14, 15);
}

#if defined(__has_builtin)
#if __has_builtin(__builtin_amdgcn_global_load_async_to_lds_b128) && \
    __has_builtin(__builtin_amdgcn_s_wait_asynccnt)
#define USE_ASYNC_LDS 1
#endif
#endif
#ifndef USE_ASYNC_LDS
#define USE_ASYNC_LDS 0
#endif

// copy 16 bytes global -> LDS (async engine when available)
__device__ __forceinline__ void cp16(_Float16* lds_dst, const _Float16* gsrc) {
#if USE_ASYNC_LDS
  __builtin_amdgcn_global_load_async_to_lds_b128(
      (__attribute__((address_space(1))) v4i*)gsrc,
      (__attribute__((address_space(3))) v4i*)lds_dst, 0, 0);
#else
  *(h8*)lds_dst = *(const h8*)gsrc;
#endif
}
template <int N>
__device__ __forceinline__ void wait_async_n() {
#if USE_ASYNC_LDS
  __builtin_amdgcn_s_wait_asynccnt(N);
#endif
}

// ============================================================================
// GEMM: C[M,N] = A[M,K](f16,row-major) @ Bt[N,K](f16, pre-transposed weights)
// block = 256 threads = 8 waves; block tile 256x64; wave tile 32x64.
// Double-buffered async staging: 5 async b128/thread/tile, wait<=5 keeps the
// prefetched tile in flight while WMMAs run on the resident tile.
// M%256==0, N%64==0, K%32==0 at all call sites.
// ============================================================================
__device__ __forceinline__ void gemm_stage(_Float16 (*As)[40], _Float16 (*Bs)[40],
                                           const _Float16* A, const _Float16* Bt,
                                           int tile_m, int tile_n, int K, int k0,
                                           int tid)
{
  #pragma unroll
  for (int i = 0; i < 4; ++i) {
    int idx = tid + i * 256;
    int r   = idx >> 2;
    int c8  = (idx & 3) * 8;
    cp16(&As[r][c8], A + (size_t)(tile_m + r) * K + k0 + c8);
  }
  int n2 = tid >> 2;
  int c8 = (tid & 3) * 8;
  cp16(&Bs[n2][c8], Bt + (size_t)(tile_n + n2) * K + k0 + c8);
}

__global__ void gemm_wmma_kernel(const _Float16* __restrict__ A,
                                 const _Float16* __restrict__ Bt,
                                 float* __restrict__ Cf,
                                 _Float16* __restrict__ Ch,
                                 const float* __restrict__ bias,
                                 const float* __restrict__ resid,
                                 int M, int N, int K, int relu)
{
  __shared__ _Float16 As16[2][256][40];  // 2 x (256 rows x 32 k, pad 8)
  __shared__ _Float16 BsT[2][64][40];    // 2 x (64 cols x 32 k, pad 8)

  const int tile_n = blockIdx.x * 64;
  const int tile_m = blockIdx.y * 256;
  const int tid  = threadIdx.x;
  const int wave = tid >> 5;
  const int lane = tid & 31;

  const int am   = lane & 15;
  const int koff = (lane >> 4) * 8;    // A frag: lanes 16-31 start at K+8
  const int bn   = lane & 15;
  const int bko  = (lane >> 4) * 8;    // B frag: lanes 16-31 start at K+8

  v8f acc[2][4] = {};

  gemm_stage(As16[0], BsT[0], A, Bt, tile_m, tile_n, K, 0, tid);

  for (int k0 = 0; k0 < K; k0 += 32) {
    const int p = (k0 >> 5) & 1;
    const bool more = (k0 + 32) < K;
    if (more) {
      gemm_stage(As16[p ^ 1], BsT[p ^ 1], A, Bt, tile_m, tile_n, K, k0 + 32, tid);
      wait_async_n<5>();   // current tile resident; prefetch stays in flight
    } else {
      wait_async_n<0>();
    }
    __syncthreads();

    v16h af[2];
    #pragma unroll
    for (int rs = 0; rs < 2; ++rs) {
      const _Float16* ar = &As16[p][wave * 32 + rs * 16 + am][0];
      af[rs] = frag_cat(*(const h8*)(ar + koff), *(const h8*)(ar + 16 + koff));
    }
    #pragma unroll
    for (int nt = 0; nt < 4; ++nt) {
      const _Float16* br = &BsT[p][nt * 16 + bn][bko];
      v16h bf = frag_cat(*(const h8*)br, *(const h8*)(br + 16));
      acc[0][nt] = wmma16(af[0], bf, acc[0][nt]);
      acc[1][nt] = wmma16(af[1], bf, acc[1][nt]);
    }
    __syncthreads();
  }

  const int cm = (lane >> 4) * 8;
  const int cn = lane & 15;
  #pragma unroll
  for (int rs = 0; rs < 2; ++rs)
    #pragma unroll
    for (int nt = 0; nt < 4; ++nt)
      #pragma unroll
      for (int r = 0; r < 8; ++r) {
        int row = tile_m + wave * 32 + rs * 16 + cm + r;
        int col = tile_n + nt * 16 + cn;
        float v = acc[rs][nt][r];
        if (bias)  v += bias[col];
        if (resid) v += resid[(size_t)row * N + col];
        if (relu)  v = fmaxf(v, 0.0f);
        if (Cf) Cf[(size_t)row * N + col] = v;
        if (Ch) Ch[(size_t)row * N + col] = (_Float16)v;
      }
}

// ============================================================================
// Fused relative-position flash attention (f16 operands, f32 softmax state).
// grid = (T/128, H, B); block = 256 = 8 waves; wave owns 16 queries.
// K/R chunks double-buffered via async copies (wait<=2); V staged transposed.
// ============================================================================
__global__ void attn_kernel(const _Float16* __restrict__ q,
                            const _Float16* __restrict__ kmem,
                            const _Float16* __restrict__ kx,
                            const _Float16* __restrict__ vmem,
                            const _Float16* __restrict__ vx,
                            const _Float16* __restrict__ rbuf,
                            const float* __restrict__ cbias,
                            const float* __restrict__ pbias,
                            _Float16* __restrict__ outp)
{
  __shared__ _Float16 kt[2][32][72];   // 32 s-rows x 64 head-dims (pad 8)
  __shared__ _Float16 rt[2][32][72];
  __shared__ _Float16 vtT[2][64][40];  // 64 head-dims x 32 s (pad 8)
  __shared__ float    sbuf[8][16][33];
  __shared__ _Float16 p16[8][16][40];
  __shared__ float    mstat[8][16], lstat[8][16], cstat[8][16];

  const int b  = blockIdx.z;
  const int h  = blockIdx.y;
  const int q0 = blockIdx.x * 128;
  const int tid = threadIdx.x, wave = tid >> 5, lane = tid & 31;
  const int qw = q0 + wave * 16;
  const int hc = h * 64;

  const int am   = lane & 15;
  const int koff = (lane >> 4) * 8;
  const int bn   = lane & 15;
  const int bko  = (lane >> 4) * 8;

  // per-thread staging coordinates (32 rows x 64 halves = 256 16B-chunks)
  const int srr = tid >> 3;
  const int sc8 = (tid & 7) * 8;

  auto stage_chunk = [&](int p, int s0) {
    const _Float16* ksrc = (s0 < 1024)
        ? (kmem + ((size_t)(b * 1024 + s0)) * 512 + hc)
        : (kx   + ((size_t)(b * 1024 + s0 - 1024)) * 512 + hc);
    const _Float16* vsrc = (s0 < 1024)
        ? (vmem + ((size_t)(b * 1024 + s0)) * 512 + hc)
        : (vx   + ((size_t)(b * 1024 + s0 - 1024)) * 512 + hc);
    const _Float16* rsrc = rbuf + (size_t)s0 * 512 + hc;
    cp16(&kt[p][srr][sc8], ksrc + (size_t)srr * 512 + sc8);
    cp16(&rt[p][srr][sc8], rsrc + (size_t)srr * 512 + sc8);
    // V transposed: one h8 load + 8 scalar LDS stores
    h8 vv = *(const h8*)(vsrc + (size_t)srr * 512 + sc8);
    #pragma unroll
    for (int j = 0; j < 8; ++j) vtT[p][sc8 + j][srr] = vv[j];
  };

  // (q + content_bias) / (q + position_bias) A-fragments over head dim 64
  const _Float16* qrow = q + ((size_t)(b * 1024 + qw + am)) * 512 + hc;
  const float* cb = cbias + hc;
  const float* pb = pbias + hc;
  v16h qc[2], qp[2];
  #pragma unroll
  for (int kk = 0; kk < 2; ++kk) {
    #pragma unroll
    for (int i = 0; i < 8; ++i) {
      int kb = kk * 32 + ((i & 4) ? 16 : 0) + koff + (i & 3) * 2;
      float f0 = (float)qrow[kb], f1 = (float)qrow[kb + 1];
      qc[kk][2 * i]     = (_Float16)(f0 + cb[kb]);
      qc[kk][2 * i + 1] = (_Float16)(f1 + cb[kb + 1]);
      qp[kk][2 * i]     = (_Float16)(f0 + pb[kb]);
      qp[kk][2 * i + 1] = (_Float16)(f1 + pb[kb + 1]);
    }
  }

  if (lane < 16) {
    mstat[wave][lane] = -1e30f;
    lstat[wave][lane] = 0.0f;
  }
  v8f oacc[4] = {};
  const float scale = 0.125f;

  stage_chunk(0, 0);

  for (int s0 = 0; s0 < 2048; s0 += 32) {
    const int p = (s0 >> 5) & 1;
    const bool more = (s0 + 32) < 2048;
    if (more) {
      stage_chunk(p ^ 1, s0 + 32);
      wait_async_n<2>();
    } else {
      wait_async_n<0>();
    }
    __syncthreads();

    // scores: (q+cb)@k^T + (q+pb)@r^T  (two 16-col tiles)
    #pragma unroll
    for (int t2 = 0; t2 < 2; ++t2) {
      v8f sc = {};
      #pragma unroll
      for (int kk = 0; kk < 2; ++kk) {
        const _Float16* kr  = &kt[p][t2 * 16 + bn][kk * 32 + bko];
        const _Float16* rr2 = &rt[p][t2 * 16 + bn][kk * 32 + bko];
        v16h bk = frag_cat(*(const h8*)kr, *(const h8*)(kr + 16));
        v16h br = frag_cat(*(const h8*)rr2, *(const h8*)(rr2 + 16));
        sc = wmma16(qc[kk], bk, sc);
        sc = wmma16(qp[kk], br, sc);
      }
      int cm = (lane >> 4) * 8;
      #pragma unroll
      for (int r = 0; r < 8; ++r)
        sbuf[wave][cm + r][t2 * 16 + (lane & 15)] = sc[r] * scale;
    }
    __syncthreads();

    // online softmax: lanes 0..15 own one query row; write probs as f16
    if (lane < 16) {
      float mo = mstat[wave][lane];
      float tmax = -1e30f;
      #pragma unroll
      for (int c = 0; c < 32; ++c) tmax = fmaxf(tmax, sbuf[wave][lane][c]);
      float mn = fmaxf(mo, tmax);
      float co = expf(mo - mn);
      float ssum = 0.0f;
      #pragma unroll
      for (int c = 0; c < 32; ++c) {
        float pexp = expf(sbuf[wave][lane][c] - mn);
        p16[wave][lane][c] = (_Float16)pexp;
        ssum += pexp;
      }
      mstat[wave][lane] = mn;
      lstat[wave][lane] = lstat[wave][lane] * co + ssum;
      cstat[wave][lane] = co;
    }
    __syncthreads();

    // rescale and p[16x32] @ v[32x64]
    {
      int cm = (lane >> 4) * 8;
      #pragma unroll
      for (int nt = 0; nt < 4; ++nt)
        #pragma unroll
        for (int r = 0; r < 8; ++r)
          oacc[nt][r] *= cstat[wave][cm + r];
    }
    const _Float16* pr = &p16[wave][am][koff];
    v16h pf = frag_cat(*(const h8*)pr, *(const h8*)(pr + 16));
    #pragma unroll
    for (int nt = 0; nt < 4; ++nt) {
      const _Float16* vr = &vtT[p][nt * 16 + bn][bko];
      v16h bv = frag_cat(*(const h8*)vr, *(const h8*)(vr + 16));
      oacc[nt] = wmma16(pf, bv, oacc[nt]);
    }
    __syncthreads();
  }

  const int cm = (lane >> 4) * 8;
  const int cn = lane & 15;
  #pragma unroll
  for (int nt = 0; nt < 4; ++nt)
    #pragma unroll
    for (int r = 0; r < 8; ++r) {
      int row = qw + cm + r;
      float v = oacc[nt][r] / lstat[wave][cm + r];
      outp[((size_t)(b * 1024 + row)) * 512 + hc + nt * 16 + cn] = (_Float16)v;
    }
}

// ============================================================================
// Elementwise helpers
// ============================================================================
__global__ void cvt_f16_kernel(const float* __restrict__ src,
                               _Float16* __restrict__ dst, int n)
{
  int i = blockIdx.x * blockDim.x + threadIdx.x;
  if (i < n) dst[i] = (_Float16)src[i];
}

// src[K][N] f32 -> dst[N][K] f16 (weight pre-transpose)
__global__ void cvt_f16_t_kernel(const float* __restrict__ src,
                                 _Float16* __restrict__ dst, int K, int N)
{
  int i = blockIdx.x * blockDim.x + threadIdx.x;
  if (i < K * N) {
    int k = i / N, n2 = i - k * N;
    dst[(size_t)n2 * K + k] = (_Float16)src[i];
  }
}

__global__ void step_embed_kernel(const float* __restrict__ cur,
                                  const float* __restrict__ wstep,
                                  float* __restrict__ xf,
                                  _Float16* __restrict__ xh,
                                  float* __restrict__ newmem)
{
  int i = blockIdx.x * blockDim.x + threadIdx.x;  // total = 4096*512
  float v = cur[i] + wstep[i & 511];
  xf[i] = v;
  xh[i] = (_Float16)v;
  newmem[i] = v;
}

__global__ void pe_kernel(_Float16* __restrict__ pe)
{
  int i = blockIdx.x * blockDim.x + threadIdx.x;  // total = 2048*512
  int s = i >> 9;
  int c = i & 511;
  float pos = (float)(2047 - s);
  int half = c >> 1;
  float inv = powf(10000.0f, -((float)(2 * half)) / 512.0f);
  float ang = pos * inv;
  pe[i] = (_Float16)((c & 1) ? cosf(ang) : sinf(ang));
}

__global__ void ln_kernel(const float* __restrict__ src,
                          const float* __restrict__ g,
                          const float* __restrict__ bta,
                          const float* __restrict__ npmask,
                          float* __restrict__ dstf,
                          _Float16* __restrict__ dsth)
{
  const int N = 512;
  int row = blockIdx.x;
  int tid = threadIdx.x;
  const float* x = src + (size_t)row * N;
  float s = 0.0f, s2 = 0.0f;
  for (int c = tid; c < N; c += 256) {
    float v = x[c];
    s += v; s2 += v * v;
  }
  __shared__ float red[512];
  red[tid] = s;
  red[256 + tid] = s2;
  __syncthreads();
  for (int off = 128; off > 0; off >>= 1) {
    if (tid < off) {
      red[tid] += red[tid + off];
      red[256 + tid] += red[256 + tid + off];
    }
    __syncthreads();
  }
  float mean = red[0] / (float)N;
  float var  = red[256] / (float)N - mean * mean;
  float rstd = rsqrtf(var + 1e-5f);
  float mk = npmask[row];
  for (int c = tid; c < N; c += 256) {
    float v = (g[c] * (x[c] - mean) * rstd + bta[c]) * mk;
    if (dstf) dstf[(size_t)row * N + c] = v;
    if (dsth) dsth[(size_t)row * N + c] = (_Float16)v;
  }
}

// ============================================================================
// Host orchestration
// ============================================================================
static inline void launch_gemm(const _Float16* A, const _Float16* Bt,
                               float* Cf, _Float16* Ch,
                               const float* bias, const float* resid,
                               int M, int N, int K, int relu, hipStream_t s)
{
  dim3 grid(N / 64, M / 256);
  gemm_wmma_kernel<<<grid, 256, 0, s>>>(A, Bt, Cf, Ch, bias, resid, M, N, K, relu);
}

extern "C" void kernel_launch(void* const* d_in, const int* in_sizes, int n_in,
                              void* d_out, int out_size, void* d_ws, size_t ws_size,
                              hipStream_t stream)
{
  (void)in_sizes; (void)n_in; (void)out_size; (void)ws_size;

  const float* layer_input = (const float*)d_in[0];
  const float* condition   = (const float*)d_in[1];
  const float* memory      = (const float*)d_in[2];
  const float* cbias       = (const float*)d_in[3];
  const float* pbias       = (const float*)d_in[4];
  const float* npmask      = (const float*)d_in[5];
  // d_in[6] slf_attn_mask: all-false -> no-op
  const float* w_step = (const float*)d_in[7];
  const float* w_cond = (const float*)d_in[8];
  const float* b_cond = (const float*)d_in[9];
  const float* wq = (const float*)d_in[10];
  const float* wk = (const float*)d_in[11];
  const float* wv = (const float*)d_in[12];
  const float* wr = (const float*)d_in[13];
  const float* wo = (const float*)d_in[14];
  const float* bo = (const float*)d_in[15];
  const float* ln1_g = (const float*)d_in[16];
  const float* ln1_b = (const float*)d_in[17];
  const float* w1 = (const float*)d_in[18];
  const float* b1 = (const float*)d_in[19];
  const float* w2 = (const float*)d_in[20];
  const float* b2 = (const float*)d_in[21];
  const float* ln2_g = (const float*)d_in[22];
  const float* ln2_b = (const float*)d_in[23];

  const size_t SEG = 2097152;  // 4096 * 512
  char* wsb = (char*)d_ws;
  size_t off = 0;
  auto alloc_f = [&](size_t n) { float* p = (float*)(wsb + off);    off += n * 4; return p; };
  auto alloc_h = [&](size_t n) { _Float16* p = (_Float16*)(wsb + off); off += n * 2;
                                 off = (off + 255) & ~(size_t)255; return p; };

  float*    cur  = alloc_f(SEG);
  float*    xb   = alloc_f(SEG);
  float*    tmp  = alloc_f(SEG);
  float*    o1f  = alloc_f(SEG);
  _Float16* xh   = alloc_h(SEG);
  _Float16* qh   = alloc_h(SEG);
  _Float16* kxh  = alloc_h(SEG);
  _Float16* kmh  = alloc_h(SEG);
  _Float16* vxh  = alloc_h(SEG);
  _Float16* vmh  = alloc_h(SEG);
  _Float16* atth = alloc_h(SEG);
  _Float16* o1h  = alloc_h(SEG);
  _Float16* hbh  = alloc_h(4096ull * 2048);
  _Float16* rh   = alloc_h(2048ull * 512);
  _Float16* peh  = alloc_h(2048ull * 512);
  _Float16* condh = alloc_h(4096ull * 64);
  _Float16* memh = alloc_h(4ull * SEG);
  _Float16* wcT  = alloc_h(512ull * 64);
  _Float16* wqT  = alloc_h(512ull * 512);
  _Float16* wkT  = alloc_h(512ull * 512);
  _Float16* wvT  = alloc_h(512ull * 512);
  _Float16* wrT  = alloc_h(512ull * 512);
  _Float16* woT  = alloc_h(512ull * 512);
  _Float16* w1T  = alloc_h(2048ull * 512);
  _Float16* w2T  = alloc_h(512ull * 2048);

  float* outp = (float*)d_out;

  // one-time conversions / pre-transposed f16 weights
  cvt_f16_t_kernel<<<(512 * 64 + 255) / 256, 256, 0, stream>>>(w_cond, wcT, 64, 512);
  cvt_f16_t_kernel<<<(512 * 512 + 255) / 256, 256, 0, stream>>>(wq, wqT, 512, 512);
  cvt_f16_t_kernel<<<(512 * 512 + 255) / 256, 256, 0, stream>>>(wk, wkT, 512, 512);
  cvt_f16_t_kernel<<<(512 * 512 + 255) / 256, 256, 0, stream>>>(wv, wvT, 512, 512);
  cvt_f16_t_kernel<<<(512 * 512 + 255) / 256, 256, 0, stream>>>(wr, wrT, 512, 512);
  cvt_f16_t_kernel<<<(512 * 512 + 255) / 256, 256, 0, stream>>>(wo, woT, 512, 512);
  cvt_f16_t_kernel<<<(512 * 2048 + 255) / 256, 256, 0, stream>>>(w1, w1T, 512, 2048);
  cvt_f16_t_kernel<<<(2048 * 512 + 255) / 256, 256, 0, stream>>>(w2, w2T, 2048, 512);
  cvt_f16_kernel<<<(4096 * 64 + 255) / 256, 256, 0, stream>>>(condition, condh, 4096 * 64);
  cvt_f16_kernel<<<(4 * 2097152 + 255) / 256, 256, 0, stream>>>(memory, memh, 4 * 2097152);

  // position encodings + R projection (step-invariant)
  pe_kernel<<<(2048 * 512) / 256, 256, 0, stream>>>(peh);
  launch_gemm(peh, wrT, nullptr, rh, nullptr, nullptr, 2048, 512, 512, 0, stream);

  // cur = layer_input + condition @ w_cond + b_cond
  launch_gemm(condh, wcT, cur, nullptr, b_cond, layer_input, 4096, 512, 64, 0, stream);

  for (int step = 0; step < 4; ++step) {
    step_embed_kernel<<<SEG / 256, 256, 0, stream>>>(
        cur, w_step + step * 512, xb, xh, outp + SEG * (size_t)(1 + step));

    const _Float16* mem_s = memh + (size_t)step * SEG;

    launch_gemm(xh,    wqT, nullptr, qh,  nullptr, nullptr, 4096, 512, 512, 0, stream);
    launch_gemm(xh,    wkT, nullptr, kxh, nullptr, nullptr, 4096, 512, 512, 0, stream);
    launch_gemm(mem_s, wkT, nullptr, kmh, nullptr, nullptr, 4096, 512, 512, 0, stream);
    launch_gemm(xh,    wvT, nullptr, vxh, nullptr, nullptr, 4096, 512, 512, 0, stream);
    launch_gemm(mem_s, wvT, nullptr, vmh, nullptr, nullptr, 4096, 512, 512, 0, stream);

    attn_kernel<<<dim3(8, 8, 4), 256, 0, stream>>>(qh, kmh, kxh, vmh, vxh, rh,
                                                   cbias, pbias, atth);

    // tmp = xb + att @ wo + bo ; (o1f, o1h) = LN1(tmp) * mask
    launch_gemm(atth, woT, tmp, nullptr, bo, xb, 4096, 512, 512, 0, stream);
    ln_kernel<<<4096, 256, 0, stream>>>(tmp, ln1_g, ln1_b, npmask, o1f, o1h);

    // FFN: hbh = relu(o1@w1+b1) f16 ; tmp = o1f + hb@w2 + b2 ; LN2
    launch_gemm(o1h, w1T, nullptr, hbh, b1, nullptr, 4096, 2048, 512, 1, stream);
    launch_gemm(hbh, w2T, tmp, nullptr, b2, o1f, 4096, 512, 2048, 0, stream);

    float* dst = (step == 3) ? outp : cur;
    ln_kernel<<<4096, 256, 0, stream>>>(tmp, ln2_g, ln2_b, npmask, dst, nullptr);
  }
}
